// PatchedQWenAttention_59897613910431
// MI455X (gfx1250) — compile-verified, matrix operand order checked by reading, not run on hardware
//
#include <hip/hip_runtime.h>
#include <hip/hip_bf16.h>

// ---------------- problem geometry (compile-time) ----------------
#define B_    4
#define SQ_   1024
#define HIST_ 1024
#define H_    32
#define DH_   128
#define BS_   64
#define NBLK_ 32
#define D_    4096
#define SKV_  2048
#define T_    4096   // B_*SQ_

// flip to 0 if the assembler rejects the async mnemonic
#define USE_ASYNC_COPY 1

typedef __attribute__((ext_vector_type(16))) _Float16 v16h;
typedef __attribute__((ext_vector_type(8)))  float    v8f;

struct __align__(16) U128 { unsigned int x, y, z, w; };
union AF { v16h h; U128 u[2]; };

__device__ __forceinline__ v8f wmma16(v16h a, v16h b, v8f c) {
  // D = A(16x32 f16) * B(32x16 f16) + C(16x16 f32)
  return __builtin_amdgcn_wmma_f32_16x16x32_f16(false, a, false, b, (short)0, c,
                                                false, false);
}

// 16-byte global -> LDS copy. Async path uses the gfx1250 ASYNCcnt-tracked
// DMA (no VGPR staging); LDS address = low 32 bits of the generic pointer
// (flat LDS aperture keeps the LDS offset in addr[31:0]).
__device__ __forceinline__ void gl2lds_b128(const void* g, void* l) {
#if USE_ASYNC_COPY
  unsigned lo = (unsigned)(unsigned long long)(uintptr_t)l;
  asm volatile("global_load_async_to_lds_b128 %0, %1, off"
               :: "v"(lo), "v"((unsigned long long)(uintptr_t)g)
               : "memory");
#else
  *(U128*)l = *(const U128*)g;
#endif
}
__device__ __forceinline__ void async_fence() {
#if USE_ASYNC_COPY
  asm volatile("s_wait_asynccnt 0x0" ::: "memory");
#endif
}

// ---------------- fp32 -> fp16 conversion (row-major) ----------------
__global__ __launch_bounds__(256) void cvt_f32_f16(const float* __restrict__ s,
                                                   _Float16* __restrict__ d,
                                                   long n) {
  long i = (long)blockIdx.x * blockDim.x + threadIdx.x;
  long stride = (long)gridDim.x * blockDim.x;
  for (; i < n; i += stride) d[i] = (_Float16)s[i];
}

// ---------------- fp32 [K,N] -> fp16 [N,K] transpose conversion -------------
__global__ __launch_bounds__(256) void cvt_f32_f16_T(
    const float* __restrict__ src, _Float16* __restrict__ dst, int K, int N) {
  __shared__ float tile[32][33];
  const int n0 = blockIdx.x * 32, k0 = blockIdx.y * 32;
  const int tx = threadIdx.x & 31, ty = threadIdx.x >> 5;   // 32 x 8
#pragma unroll
  for (int j = 0; j < 32; j += 8)
    tile[ty + j][tx] = src[(size_t)(k0 + ty + j) * N + n0 + tx];
  __syncthreads();
#pragma unroll
  for (int j = 0; j < 32; j += 8)
    dst[(size_t)(n0 + ty + j) * K + k0 + tx] = (_Float16)tile[tx][ty + j];
}

// ---------------- WMMA GEMM: C[M,N] = A[M,K] * Bt[N,K]^T (+bias) ------------
// 256 threads (8 waves). Block tile 128x64, BK=64, double-buffered LDS filled
// by async global->LDS copies. Wave (wm,wn) owns a 32x32 output tile:
// 4 f32 accumulators, 8 WMMAs per K-stage, one barrier per stage.
template <bool OUT_HALF>
__global__ __launch_bounds__(256) void gemm_wmma(
    const _Float16* __restrict__ A,   // [M,K] row-major
    const _Float16* __restrict__ Bt,  // [N,K] row-major (B transposed)
    const float* __restrict__ bias, float* __restrict__ Cf,
    _Float16* __restrict__ Ch, int M, int N, int K) {
  __shared__ __align__(16) _Float16 As[2][128][64];   // 32 KB
  __shared__ __align__(16) _Float16 Bs[2][64][64];    // 16 KB

  const int t    = threadIdx.x;
  const int lane = t & 31;
  const int wave = t >> 5;
  const int wm   = wave & 3;     // 0..3 -> 32-row strip
  const int wn   = wave >> 2;    // 0..1 -> 32-col strip
  const int ln   = lane & 15;
  const int kb   = (lane < 16) ? 0 : 8;   // K sub-chunk base per ISA layout
  const int m0   = blockIdx.y * 128;
  const int n0   = blockIdx.x * 64;

  v8f acc[2][2];
#pragma unroll
  for (int i = 0; i < 2; ++i)
#pragma unroll
    for (int j = 0; j < 2; ++j)
#pragma unroll
      for (int r = 0; r < 8; ++r) acc[i][j][r] = 0.f;

  auto load_stage = [&](int k0, int buf) {
#pragma unroll
    for (int i = 0; i < 4; ++i) {            // A: 128x64 halves = 1024 x b128
      int u = t + i * 256;
      int m = u >> 3, kc = (u & 7) * 8;
      gl2lds_b128(&A[(size_t)(m0 + m) * K + k0 + kc], &As[buf][m][kc]);
    }
#pragma unroll
    for (int i = 0; i < 2; ++i) {            // B: 64x64 halves = 512 x b128
      int u = t + i * 256;
      int n = u >> 3, kc = (u & 7) * 8;
      gl2lds_b128(&Bt[(size_t)(n0 + n) * K + k0 + kc], &Bs[buf][n][kc]);
    }
  };

  load_stage(0, 0);
  async_fence();
  __syncthreads();

  const int nstage = K >> 6;
  for (int s = 0; s < nstage; ++s) {
    const int cur = s & 1;
    if (s + 1 < nstage) load_stage((s + 1) << 6, cur ^ 1);  // prefetch next

#pragma unroll
    for (int c = 0; c < 2; ++c) {            // two 32-wide K chunks per stage
      AF a0, a1, b0, b1;
      const _Float16* ap0 = &As[cur][wm * 32 + ln][c * 32 + kb];
      const _Float16* ap1 = &As[cur][wm * 32 + 16 + ln][c * 32 + kb];
      const _Float16* bp0 = &Bs[cur][wn * 32 + ln][c * 32 + kb];
      const _Float16* bp1 = &Bs[cur][wn * 32 + 16 + ln][c * 32 + kb];
      a0.u[0] = *(const U128*)ap0; a0.u[1] = *(const U128*)(ap0 + 16);
      a1.u[0] = *(const U128*)ap1; a1.u[1] = *(const U128*)(ap1 + 16);
      b0.u[0] = *(const U128*)bp0; b0.u[1] = *(const U128*)(bp0 + 16);
      b1.u[0] = *(const U128*)bp1; b1.u[1] = *(const U128*)(bp1 + 16);
      acc[0][0] = wmma16(a0.h, b0.h, acc[0][0]);
      acc[0][1] = wmma16(a0.h, b1.h, acc[0][1]);
      acc[1][0] = wmma16(a1.h, b0.h, acc[1][0]);
      acc[1][1] = wmma16(a1.h, b1.h, acc[1][1]);
    }
    async_fence();
    __syncthreads();
  }

  // C layout: lanes 0-15 -> rows r, lanes 16-31 -> rows r+8 ; col = ln
  const int rowoff = (lane < 16) ? 0 : 8;
#pragma unroll
  for (int i = 0; i < 2; ++i)
#pragma unroll
    for (int j = 0; j < 2; ++j) {
      const int rbase = m0 + wm * 32 + i * 16 + rowoff;
      const int col = n0 + wn * 32 + j * 16 + ln;
      const float badd = bias ? bias[col] : 0.f;
#pragma unroll
      for (int r = 0; r < 8; ++r) {
        float v = acc[i][j][r] + badd;
        size_t o = (size_t)(rbase + r) * N + col;
        if (OUT_HALF) Ch[o] = (_Float16)v;
        else          Cf[o] = v;
      }
    }
}

// ---------------- QKV postprocess: RoPE + logn + KV gather ------------------
__global__ __launch_bounds__(256) void build_qkv(
    const _Float16* __restrict__ qkv,      // [T, 3D] biased
    const float* __restrict__ cosT, const float* __restrict__ sinT,
    const float* __restrict__ kcache, const float* __restrict__ vcache,
    const int* __restrict__ blockOff,      // [B, NBLK]
    const long long* __restrict__ posIds,  // [T]
    _Float16* __restrict__ Qo,             // [B,H,SQ,DH]
    _Float16* __restrict__ Ko,             // [B,H,SKV,DH]
    _Float16* __restrict__ VTo) {          // [B,H,DH,SKV]
  size_t g = (size_t)blockIdx.x * blockDim.x + threadIdx.x;
  int d   = (int)(g & (DH_ - 1));
  int h   = (int)((g >> 7) & (H_ - 1));
  int pos = (int)((g >> 12) & (SKV_ - 1));
  int b   = (int)(g >> 23);
  if (b >= B_) return;
  size_t bh = (size_t)(b * H_ + h);

  if (pos < HIST_) {
    int blk = blockOff[b * NBLK_ + (pos >> 6)];
    size_t src = (((size_t)blk * BS_ + (pos & (BS_ - 1))) * H_ + h) * DH_ + d;
    Ko[(bh * SKV_ + pos) * DH_ + d] = (_Float16)kcache[src];
    VTo[(bh * DH_ + d) * SKV_ + pos] = (_Float16)vcache[src];
  } else {
    int q = pos - HIST_;
    int t = b * SQ_ + q;
    long long pid = posIds[t];
    float c = cosT[(size_t)pid * DH_ + d];
    float s = sinT[(size_t)pid * DH_ + d];
    const _Float16* base = qkv + (size_t)t * (3 * D_);
    int hd = h * DH_ + d;
    int dp = (d < 64) ? d + 64 : d - 64;
    float sgn = (d < 64) ? -1.f : 1.f;
    float qv = (float)base[hd];
    float qp = (float)base[h * DH_ + dp];
    float kv = (float)base[D_ + hd];
    float kp = (float)base[D_ + h * DH_ + dp];
    float vv = (float)base[2 * D_ + hd];
    float qr = qv * c + sgn * qp * s;
    float kr = kv * c + sgn * kp * s;
    float p1 = (float)(pid + 1);
    float logn = (p1 > 1024.f) ? (__logf(p1) * (1.0f / 6.931471805599453f)) : 1.f;
    qr *= logn * 0.08838834764831845f;   // fold logn and 1/sqrt(DH) into Q
    Qo[(bh * SQ_ + q) * DH_ + d] = (_Float16)qr;
    Ko[(bh * SKV_ + pos) * DH_ + d] = (_Float16)kr;
    VTo[(bh * DH_ + d) * SKV_ + pos] = (_Float16)vv;
  }
}

// ---------------- flash attention: one wave per 16-row Q tile ---------------
__global__ __launch_bounds__(128) void attn_wmma(
    const _Float16* __restrict__ Q,    // [B,H,SQ,DH] (scaled)
    const _Float16* __restrict__ K,    // [B,H,SKV,DH]
    const _Float16* __restrict__ VT,   // [B,H,DH,SKV]
    _Float16* __restrict__ attn) {     // [T, D]
  __shared__ __align__(16) _Float16 Pl[4][16][32];  // per-wave P transpose tile

  const int lane = threadIdx.x & 31;
  const int wave = threadIdx.x >> 5;
  const int bh = blockIdx.x;           // b*H + h
  const int b = bh >> 5;               // H_ == 32
  const int h = bh & 31;
  const int q0 = (blockIdx.y * 4 + wave) * 16;
  const int ln = lane & 15;
  const int kb = (lane < 16) ? 0 : 8;
  const int rowoff = (lane < 16) ? 0 : 8;

  const _Float16* Qb = Q + ((size_t)bh * SQ_ + q0) * DH_;
  const _Float16* Kb = K + (size_t)bh * SKV_ * DH_;
  const _Float16* Vb = VT + (size_t)bh * DH_ * SKV_;

  AF qa[4];
#pragma unroll
  for (int c = 0; c < 4; ++c) {
    const _Float16* p = Qb + ln * DH_ + c * 32 + kb;
    qa[c].u[0] = *(const U128*)p;
    qa[c].u[1] = *(const U128*)(p + 16);
  }

  v8f out[8];
  float mrow[8], lrow[8];
#pragma unroll
  for (int d = 0; d < 8; ++d) {
#pragma unroll
    for (int r = 0; r < 8; ++r) out[d][r] = 0.f;
  }
#pragma unroll
  for (int r = 0; r < 8; ++r) { mrow[r] = -1e30f; lrow[r] = 0.f; }

  const int qabs = HIST_ + q0;
  const int kend = qabs + 16;

  for (int k0 = 0; k0 < kend; k0 += 32) {
    v8f s0, s1;
#pragma unroll
    for (int r = 0; r < 8; ++r) { s0[r] = 0.f; s1[r] = 0.f; }
#pragma unroll
    for (int c = 0; c < 4; ++c) {
      AF bk0, bk1;
      const _Float16* kp0 = Kb + (size_t)(k0 + ln) * DH_ + c * 32 + kb;
      const _Float16* kp1 = kp0 + 16 * DH_;
      bk0.u[0] = *(const U128*)kp0; bk0.u[1] = *(const U128*)(kp0 + 16);
      bk1.u[0] = *(const U128*)kp1; bk1.u[1] = *(const U128*)(kp1 + 16);
      s0 = wmma16(qa[c].h, bk0.h, s0);
      s1 = wmma16(qa[c].h, bk1.h, s1);
    }

    const bool needMask = (k0 + 31 > qabs);
    const int col0 = k0 + ln, col1 = col0 + 16;
    const int rbase = qabs + rowoff;

#pragma unroll
    for (int r = 0; r < 8; ++r) {
      float v0 = s0[r], v1 = s1[r];
      if (needMask) {
        if (col0 > rbase + r) v0 = -1e30f;
        if (col1 > rbase + r) v1 = -1e30f;
      }
      float mloc = fmaxf(v0, v1);
#pragma unroll
      for (int off = 8; off >= 1; off >>= 1)
        mloc = fmaxf(mloc, __shfl_xor(mloc, off, 32));
      float mnew = fmaxf(mrow[r], mloc);
      float al = __expf(mrow[r] - mnew);
      float p0 = __expf(v0 - mnew);
      float p1 = __expf(v1 - mnew);
      float ls = p0 + p1;
#pragma unroll
      for (int off = 8; off >= 1; off >>= 1) ls += __shfl_xor(ls, off, 32);
      lrow[r] = lrow[r] * al + ls;
      mrow[r] = mnew;
#pragma unroll
      for (int d = 0; d < 8; ++d) out[d][r] *= al;
      Pl[wave][r + rowoff][ln]      = (_Float16)p0;
      Pl[wave][r + rowoff][ln + 16] = (_Float16)p1;
    }

    AF pa;
    pa.u[0] = *(const U128*)&Pl[wave][ln][kb];
    pa.u[1] = *(const U128*)&Pl[wave][ln][kb + 16];

#pragma unroll
    for (int d = 0; d < 8; ++d) {
      AF vb;
      const _Float16* vp = Vb + (size_t)(d * 16 + ln) * SKV_ + k0 + kb;
      vb.u[0] = *(const U128*)vp;
      vb.u[1] = *(const U128*)(vp + 16);
      out[d] = wmma16(pa.h, vb.h, out[d]);
    }
  }

#pragma unroll
  for (int r = 0; r < 8; ++r) {
    float inv = 1.0f / lrow[r];
    int t = b * SQ_ + q0 + r + rowoff;
    _Float16* dst = attn + (size_t)t * D_ + h * DH_ + ln;
#pragma unroll
    for (int d = 0; d < 8; ++d) dst[d * 16] = (_Float16)(out[d][r] * inv);
  }
}

// ---------------- launch ----------------
extern "C" void kernel_launch(void* const* d_in, const int* in_sizes, int n_in,
                              void* d_out, int out_size, void* d_ws,
                              size_t ws_size, hipStream_t stream) {
  (void)in_sizes; (void)n_in; (void)out_size; (void)ws_size;
  const float* hidden = (const float*)d_in[0];   // [T, D]
  const float* w_qkv  = (const float*)d_in[1];   // [D, 3D]
  const float* b_qkv  = (const float*)d_in[2];   // [3D]
  const float* w_proj = (const float*)d_in[3];   // [D, D]
  const float* cosT   = (const float*)d_in[4];   // [4096, DH]
  const float* sinT   = (const float*)d_in[5];
  const float* kcache = (const float*)d_in[6];   // [B*NBLK, BS, H, DH]
  const float* vcache = (const float*)d_in[7];
  const int*   blkOff = (const int*)d_in[8];     // [B, NBLK]
  const long long* posIds = (const long long*)d_in[9];  // [T]
  float* out = (float*)d_out;                    // [T, D]

  // workspace partition (total 469,762,048 bytes ~ 448 MiB)
  char* ws = (char*)d_ws;
  _Float16* hiddenH = (_Float16*)(ws);              //  33,554,432
  _Float16* wqkvT   = (_Float16*)(ws + 33554432);   // 100,663,296  [3D][D]
  _Float16* wprojT  = (_Float16*)(ws + 134217728);  //  33,554,432  [D][D]
  _Float16* qkvH    = (_Float16*)(ws + 167772160);  // 100,663,296
  _Float16* QH      = (_Float16*)(ws + 268435456);  //  33,554,432
  _Float16* KH      = (_Float16*)(ws + 301989888);  //  67,108,864
  _Float16* VTH     = (_Float16*)(ws + 369098752);  //  67,108,864
  _Float16* attnH   = (_Float16*)(ws + 436207616);  //  33,554,432

  // 1) operand conversion: activations row-major, weights transposed -> [N,K]
  cvt_f32_f16<<<8192, 256, 0, stream>>>(hidden, hiddenH, (long)T_ * D_);
  cvt_f32_f16_T<<<dim3(3 * D_ / 32, D_ / 32), 256, 0, stream>>>(
      w_qkv, wqkvT, D_, 3 * D_);
  cvt_f32_f16_T<<<dim3(D_ / 32, D_ / 32), 256, 0, stream>>>(
      w_proj, wprojT, D_, D_);

  // 2) QKV projection: [T,D] x [D,3D] + bias -> f16 [T,3D]
  gemm_wmma<true><<<dim3(3 * D_ / 64, T_ / 128), 256, 0, stream>>>(
      hiddenH, wqkvT, b_qkv, nullptr, qkvH, T_, 3 * D_, D_);

  // 3) RoPE + logn + KV gather (cache history + new tokens)
  {
    size_t total = (size_t)B_ * SKV_ * H_ * DH_;   // 2^25
    build_qkv<<<(unsigned)(total / 256), 256, 0, stream>>>(
        qkvH, cosT, sinT, kcache, vcache, blkOff, posIds, QH, KH, VTH);
  }

  // 4) flash attention: grid (B*H, SQ/64), 4 waves/block, 1 q-tile/wave
  attn_wmma<<<dim3(B_ * H_, SQ_ / 64), 128, 0, stream>>>(QH, KH, VTH, attnH);

  // 5) output projection: [T,D] x [D,D] -> fp32 d_out
  gemm_wmma<false><<<dim3(D_ / 64, T_ / 128), 256, 0, stream>>>(
      attnH, wprojT, nullptr, out, nullptr, T_, D_, D_);
}
